// MultiHeadAttention_7679401525625
// MI455X (gfx1250) — compile-verified, matrix operand order checked by reading, not run on hardware
//
#include <hip/hip_runtime.h>
#include <hip/hip_bf16.h>
#include <stdint.h>

// ---------------------------------------------------------------------------
// MI455X (gfx1250) multi-head attention forward, bf16 WMMA pipeline.
//  - all GEMM-class work on v_wmma_f32_16x16x32_bf16
//  - GEMM LDS staging via GLOBAL_LOAD_ASYNC_TO_LDS_B128 (ASYNCcnt)
//  - attention K/V^T staging via TENSOR_LOAD_TO_LDS (TDM, TENSORcnt)
//  - V is written transposed ([B,H,dk,S]) at projection time so the flash
//    attention inner loop has zero transposes.
// ---------------------------------------------------------------------------

typedef __bf16 bf16;
typedef __attribute__((ext_vector_type(16))) __bf16 v16bf;
typedef __attribute__((ext_vector_type(8)))  __bf16 v8bf;
typedef __attribute__((ext_vector_type(4)))  __bf16 v4bf;
typedef __attribute__((ext_vector_type(8)))  float  v8f;
typedef __attribute__((ext_vector_type(4)))  int    v4i;
typedef __attribute__((ext_vector_type(8)))  int    v8i;
typedef __attribute__((ext_vector_type(4)))  unsigned int v4u;

#define D_MODEL 1024
#define NHEAD   16
#define DKDIM   64
#define BATCH   4
#define SEQ     2048
#define MTOT    (BATCH * SEQ)   // 8192

#define AS1 __attribute__((address_space(1)))
#define AS3 __attribute__((address_space(3)))

#if defined(__has_builtin)
#  if __has_builtin(__builtin_amdgcn_global_load_async_to_lds_b128)
#    define HAVE_ASYNC_LDS 1
#  else
#    define HAVE_ASYNC_LDS 0
#  endif
#  if __has_builtin(__builtin_amdgcn_s_wait_asynccnt)
#    define HAVE_WAIT_ASYNC 1
#  else
#    define HAVE_WAIT_ASYNC 0
#  endif
#  if __has_builtin(__builtin_amdgcn_tensor_load_to_lds) && \
      __has_builtin(__builtin_amdgcn_s_wait_tensorcnt)
#    define HAVE_TDM 1
#  else
#    define HAVE_TDM 0
#  endif
#else
#  define HAVE_ASYNC_LDS 0
#  define HAVE_WAIT_ASYNC 0
#  define HAVE_TDM 0
#endif

static __device__ __forceinline__ v8f zero8() {
  v8f z = {0.f, 0.f, 0.f, 0.f, 0.f, 0.f, 0.f, 0.f};
  return z;
}

// 16-byte global -> LDS copy; async DMA path on CDNA5, sync fallback otherwise.
static __device__ __forceinline__ void cp_b128(const bf16* g, bf16* l) {
#if HAVE_ASYNC_LDS
  __builtin_amdgcn_global_load_async_to_lds_b128(
      (AS1 v4i*)g, (AS3 v4i*)l, 0, 0);
#else
  *(v8bf*)l = *(const v8bf*)g;
#endif
}

static __device__ __forceinline__ void wait_cp() {
#if HAVE_ASYNC_LDS
#  if HAVE_WAIT_ASYNC
  __builtin_amdgcn_s_wait_asynccnt(0);
#  else
  asm volatile("s_wait_asynccnt 0x0" ::: "memory");
#  endif
#endif
}

// LDS byte address (addrspace(3) offset) of a generic pointer to shared mem.
static __device__ __forceinline__ uint32_t lds_addr_of(const void* p) {
  return (uint32_t)(uintptr_t)(AS3 const void*)p;
}

// TDM: DMA one 2D bf16 tile [tile_y rows x tile_x elems] from global (row
// stride stride_elems) into contiguous LDS at lds_byte.  D# layout per CDNA5
// ISA ch.8: group0 = {count/flags, lds_addr, global_addr, type=2}, group1 =
// {data_size=2B, tensor dims, tile dims, dim0 stride}.  Tracked by TENSORcnt.
static __device__ __forceinline__ void tdm_load_2d(const void* gptr,
                                                   uint32_t lds_byte,
                                                   uint32_t tile_x,
                                                   uint32_t tile_y,
                                                   uint32_t stride_elems) {
#if HAVE_TDM
  const uint64_t ga = (uint64_t)(uintptr_t)gptr;
  v4u g0;
  g0[0] = 1u;                                   // count=1 (valid), user flags 0
  g0[1] = lds_byte;                             // lds_addr
  g0[2] = (uint32_t)ga;                         // global_addr[31:0]
  g0[3] = (uint32_t)(ga >> 32) | 0x80000000u;   // global_addr[56:32] | type=2
  v8i g1;
  g1[0] = (int)0x00010000u;                     // data_size=1 (2 bytes)
  g1[1] = (int)((tile_x & 0xFFFFu) << 16);                    // tensor_dim0 lo
  g1[2] = (int)(((tile_x >> 16) & 0xFFFFu) |
                ((tile_y & 0xFFFFu) << 16));    // tensor_dim0 hi | tensor_dim1 lo
  g1[3] = (int)(((tile_y >> 16) & 0xFFFFu) |
                ((tile_x & 0xFFFFu) << 16));    // tensor_dim1 hi | tile_dim0
  g1[4] = (int)(tile_y & 0xFFFFu);              // tile_dim1 | tile_dim2=0
  g1[5] = (int)stride_elems;                    // tensor_dim0_stride[31:0]
  g1[6] = 0;                                    // stride hi | dim1_stride lo
  g1[7] = 0;
  v4i gz = {0, 0, 0, 0};
#  if __clang_major__ >= 23
  v8i gz8 = {0, 0, 0, 0, 0, 0, 0, 0};
  __builtin_amdgcn_tensor_load_to_lds(g0, g1, gz, gz, gz8, 0);
#  else
  __builtin_amdgcn_tensor_load_to_lds(g0, g1, gz, gz, 0);
#  endif
#endif
}

static __device__ __forceinline__ void wait_tdm() {
#if HAVE_TDM
  __builtin_amdgcn_s_wait_tensorcnt(0);
#endif
}

// Load a 16x32 bf16 WMMA fragment (A-layout / K-major-per-lane B-layout).
// lane<16: row=lane, K in {0..7,16..23}; lane>=16: row=lane-16, K in {8..15,24..31}.
static __device__ __forceinline__ v16bf frag_ld(const bf16* p, int ld) {
  const int lane = threadIdx.x & 31;
  const bf16* q = p + (lane & 15) * ld + ((lane >> 4) << 3);
  v8bf lo = *(const v8bf*)(q);
  v8bf hi = *(const v8bf*)(q + 16);
  v16bf f;
#pragma unroll
  for (int i = 0; i < 8; ++i) { f[i] = lo[i]; f[i + 8] = hi[i]; }
  return f;
}

static __device__ __forceinline__ v8f wmma_bf16(v16bf a, v16bf b, v8f c) {
  return __builtin_amdgcn_wmma_f32_16x16x32_bf16(false, a, false, b,
                                                 (short)0, c, false, false);
}

// ---------------------------------------------------------------------------
// Kernel 1: f32 -> bf16 conversion (streaming).
// ---------------------------------------------------------------------------
__global__ void mha_cvt_bf16(const float* __restrict__ src,
                             bf16* __restrict__ dst, int n) {
  int i = (blockIdx.x * blockDim.x + threadIdx.x) * 4;
  if (i >= n) return;
  const float4 f = *(const float4*)(src + i);
  v4bf o;
  o[0] = (bf16)f.x; o[1] = (bf16)f.y; o[2] = (bf16)f.z; o[3] = (bf16)f.w;
  *(v4bf*)(dst + i) = o;
}

// ---------------------------------------------------------------------------
// Kernel 2/4: C = A * B^T.  A: [8192,1024] bf16 rowmajor, B: [1024,1024] bf16
// rowmajor.  Block tile 128x128, BK=32, 8 waves x (2x4) WMMA subtiles.
// Async double-buffered LDS staging.
// MODE 0: bf16 out, [B,H,S,DK] head layout          (Q, K projections)
// MODE 1: f32 out, rowmajor [M, D_MODEL]            (output projection)
// MODE 2: bf16 out, transposed [B,H,DK,S] layout    (V projection)
// ---------------------------------------------------------------------------
template <int MODE>
__global__ __launch_bounds__(256)
void mha_gemm_bt(const bf16* __restrict__ A, const bf16* __restrict__ Bm,
                 void* __restrict__ out) {
  // 32 KB: [0..8191] = sA double buffer, [8192..16383] = sB double buffer.
  // Reused as a 128x128 bf16 output-staging buffer in the epilogue.
  __shared__ __align__(16) bf16 smem[16384];

  const int tid  = threadIdx.x;
  const int wave = tid >> 5;
  const int lane = tid & 31;
  const int m0 = blockIdx.y * 128;
  const int n0 = blockIdx.x * 128;
  const int wr = (wave >> 1) * 32;
  const int wc = (wave & 1) * 64;

  v8f acc[2][4];
#pragma unroll
  for (int i = 0; i < 2; ++i)
#pragma unroll
    for (int j = 0; j < 4; ++j) acc[i][j] = zero8();

  const int NSTEP = D_MODEL / 32;

  // stage K-step 0 into buffer 0
#pragma unroll
  for (int p = 0; p < 2; ++p) {
    int slot = tid + p * 256;               // 0..511
    int r = slot >> 2, c = (slot & 3) << 3; // 128 rows x 32 cols
    cp_b128(A  + (size_t)(m0 + r) * D_MODEL + c, &smem[r * 32 + c]);
    cp_b128(Bm + (size_t)(n0 + r) * D_MODEL + c, &smem[8192 + r * 32 + c]);
  }
  wait_cp();
  __syncthreads();

  for (int kt = 0; kt < NSTEP; ++kt) {
    const int cur = kt & 1;
    if (kt + 1 < NSTEP) {
      const int k0 = (kt + 1) * 32;
      const int nxt = cur ^ 1;   // last read in the previous iteration, before its barrier; safe to fill
#pragma unroll
      for (int p = 0; p < 2; ++p) {
        int slot = tid + p * 256;
        int r = slot >> 2, c = (slot & 3) << 3;
        cp_b128(A  + (size_t)(m0 + r) * D_MODEL + k0 + c,
                &smem[nxt * 4096 + r * 32 + c]);
        cp_b128(Bm + (size_t)(n0 + r) * D_MODEL + k0 + c,
                &smem[8192 + nxt * 4096 + r * 32 + c]);
      }
    }

    v16bf af[2], bfv[4];
#pragma unroll
    for (int i = 0; i < 2; ++i)
      af[i] = frag_ld(&smem[cur * 4096 + (wr + i * 16) * 32], 32);
#pragma unroll
    for (int j = 0; j < 4; ++j)
      bfv[j] = frag_ld(&smem[8192 + cur * 4096 + (wc + j * 16) * 32], 32);
#pragma unroll
    for (int i = 0; i < 2; ++i)
#pragma unroll
      for (int j = 0; j < 4; ++j) acc[i][j] = wmma_bf16(af[i], bfv[j], acc[i][j]);

    if (kt + 1 < NSTEP) {
      wait_cp();
      __syncthreads();
    }
  }

  // Epilogue.  C/D layout: VGPR r <-> row r (lanes 0-15) / r+8 (lanes 16-31);
  // lanes are columns.
  const int lr = lane & 15;
  const int hh = lane >> 4;

  if (MODE == 0 || MODE == 2) {
    // Bounce through LDS so global stores are contiguous b128.
    __syncthreads();                 // staging buffers are done being read
#pragma unroll
    for (int i = 0; i < 2; ++i)
#pragma unroll
      for (int j = 0; j < 4; ++j)
#pragma unroll
        for (int r = 0; r < 8; ++r)
          smem[(wr + i * 16 + r + hh * 8) * 128 + wc + j * 16 + lr] =
              (bf16)acc[i][j][r];
    __syncthreads();

    const int b = m0 >> 11, s0 = m0 & 2047;
    if (MODE == 2) {
      // V^T = [B,H,DK,S]: transposed read, rows of seq.
#pragma unroll
      for (int p = 0; p < 8; ++p) {
        int slot = tid + p * 256;          // 0..2047
        int n = slot >> 4;                 // local column (head dim)
        int sc = (slot & 15) << 3;         // local row chunk (seq)
        v8bf valv;
#pragma unroll
        for (int e = 0; e < 8; ++e) valv[e] = smem[(sc + e) * 128 + n];
        int gn = n0 + n, h = gn >> 6, d = gn & 63;
        *(v8bf*)((bf16*)out + (((size_t)b * NHEAD + h) * DKDIM + d) * SEQ +
                 s0 + sc) = valv;
      }
    } else {
      // [B,H,S,DK]: contiguous read, rows of head dim.
#pragma unroll
      for (int p = 0; p < 8; ++p) {
        int slot = tid + p * 256;          // 0..2047
        int row = slot >> 4;               // local row (seq)
        int c = (slot & 15) << 3;          // local col chunk (head dim)
        v8bf valv = *(const v8bf*)&smem[row * 128 + c];
        int gn = n0 + c, h = gn >> 6, d = gn & 63;
        *(v8bf*)((bf16*)out + (((size_t)b * NHEAD + h) * SEQ + s0 + row) *
                     DKDIM + d) = valv;
      }
    }
    return;
  }

  // MODE 1: f32 rowmajor, stores coalesce across lanes.
#pragma unroll
  for (int i = 0; i < 2; ++i)
#pragma unroll
    for (int j = 0; j < 4; ++j)
#pragma unroll
      for (int r = 0; r < 8; ++r) {
        int m = m0 + wr + i * 16 + r + hh * 8;
        int n = n0 + wc + j * 16 + lr;
        ((float*)out)[(size_t)m * D_MODEL + n] = acc[i][j][r];
      }
}

// ---------------------------------------------------------------------------
// Kernel 3: flash attention per (b,h).  Block = 4 waves, 64 query rows.
// Double-buffered K ([key][dk]) and V^T ([dk][key]) chunks staged by the
// Tensor Data Mover (wave 0 issues; TENSORcnt waits) with an async-b128
// fallback.  Online softmax with half-wave shuffle reductions.
// ---------------------------------------------------------------------------
__global__ __launch_bounds__(128)
void mha_attention(const bf16* __restrict__ Qp, const bf16* __restrict__ Kp,
                   const bf16* __restrict__ VpT, bf16* __restrict__ ctx) {
  __shared__ __align__(16) bf16 sK[2][64 * 64];   // [key][dk]
  __shared__ __align__(16) bf16 sVt[2][64 * 64];  // [dk][key]
  __shared__ __align__(16) bf16 sP[4][16 * 64];   // per-wave P/O tile [q][key]

  const int tid  = threadIdx.x;
  const int wave = tid >> 5;
  const int lane = tid & 31;
  const int lr = lane & 15, hh = lane >> 4;
  const int bh = blockIdx.y;            // b*16+h
  const int q0 = blockIdx.x * 64;

  const bf16* Qh  = Qp  + (size_t)bh * SEQ * DKDIM;
  const bf16* Kh  = Kp  + (size_t)bh * SEQ * DKDIM;
  const bf16* VhT = VpT + (size_t)bh * DKDIM * SEQ;

  // Preload + pre-scale this wave's Q fragments (16 rows x 64 dk).
  v16bf qf[2];
#pragma unroll
  for (int t = 0; t < 2; ++t) {
    v16bf f = frag_ld(Qh + (size_t)(q0 + wave * 16) * DKDIM + t * 32, DKDIM);
#pragma unroll
    for (int e = 0; e < 16; ++e) f[e] = (bf16)((float)f[e] * 0.125f); // 1/sqrt(64)
    qf[t] = f;
  }

  float mrow[8], lrow[8];
#pragma unroll
  for (int r = 0; r < 8; ++r) { mrow[r] = -3.0e30f; lrow[r] = 0.0f; }
  v8f oacc[4];
#pragma unroll
  for (int j = 0; j < 4; ++j) oacc[j] = zero8();

  const int NC = SEQ / 64;

  // stage chunk 0 into buffer 0
#if HAVE_TDM
  if (wave == 0) {
    tdm_load_2d(Kh,  lds_addr_of(&sK[0][0]),  64, 64, DKDIM);
    tdm_load_2d(VhT, lds_addr_of(&sVt[0][0]), 64, 64, SEQ);
    wait_tdm();
  }
#else
#pragma unroll
  for (int p = 0; p < 4; ++p) {
    int slot = tid + p * 128;               // 0..511
    int r = slot >> 3, c = (slot & 7) << 3; // 64 x 64
    cp_b128(Kh  + (size_t)r * DKDIM + c, &sK[0][r * 64 + c]);
    cp_b128(VhT + (size_t)r * SEQ + c,   &sVt[0][r * 64 + c]);
  }
  wait_cp();
#endif
  __syncthreads();

  for (int kc = 0; kc < NC; ++kc) {
    const int cur = kc & 1;
    if (kc + 1 < NC) {
      const int nxt = cur ^ 1;
      const int key0 = (kc + 1) * 64;
#if HAVE_TDM
      if (wave == 0) {
        tdm_load_2d(Kh + (size_t)key0 * DKDIM, lds_addr_of(&sK[nxt][0]),
                    64, 64, DKDIM);
        tdm_load_2d(VhT + key0, lds_addr_of(&sVt[nxt][0]), 64, 64, SEQ);
      }
#else
#pragma unroll
      for (int p = 0; p < 4; ++p) {
        int slot = tid + p * 128;
        int r = slot >> 3, c = (slot & 7) << 3;
        cp_b128(Kh  + (size_t)(key0 + r) * DKDIM + c, &sK[nxt][r * 64 + c]);
        cp_b128(VhT + (size_t)r * SEQ + key0 + c,     &sVt[nxt][r * 64 + c]);
      }
#endif
    }

    // scores S = Q * K^T  (16q x 64k per wave)
    v8f sc[4];
#pragma unroll
    for (int j = 0; j < 4; ++j) {
      v8f c = zero8();
#pragma unroll
      for (int t = 0; t < 2; ++t) {
        v16bf kf = frag_ld(&sK[cur][(j * 16) * 64 + t * 32], 64);
        c = wmma_bf16(qf[t], kf, c);
      }
      sc[j] = c;
    }

    // Row-wise max across 64 keys.
    float rm[8];
#pragma unroll
    for (int r = 0; r < 8; ++r) {
      float v = fmaxf(fmaxf(sc[0][r], sc[1][r]), fmaxf(sc[2][r], sc[3][r]));
#pragma unroll
      for (int m = 1; m < 16; m <<= 1) v = fmaxf(v, __shfl_xor(v, m, 32));
      rm[r] = v;
    }
    float alpha[8];
#pragma unroll
    for (int r = 0; r < 8; ++r) {
      float mn = fmaxf(mrow[r], rm[r]);
      alpha[r] = __expf(mrow[r] - mn);
      mrow[r] = mn;
    }

    // P = exp(S - m); stage to wave-private LDS; accumulate row sums.
    float rs[8];
#pragma unroll
    for (int r = 0; r < 8; ++r) rs[r] = 0.0f;
#pragma unroll
    for (int j = 0; j < 4; ++j)
#pragma unroll
      for (int r = 0; r < 8; ++r) {
        float p = __expf(sc[j][r] - mrow[r]);
        rs[r] += p;
        sP[wave][(r + hh * 8) * 64 + j * 16 + lr] = (bf16)p;
      }
#pragma unroll
    for (int r = 0; r < 8; ++r) {
#pragma unroll
      for (int m = 1; m < 16; m <<= 1) rs[r] += __shfl_xor(rs[r], m, 32);
      lrow[r] = lrow[r] * alpha[r] + rs[r];
    }

    // Rescale running O, then O += P * V.
#pragma unroll
    for (int j = 0; j < 4; ++j)
#pragma unroll
      for (int r = 0; r < 8; ++r) oacc[j][r] *= alpha[r];

    v16bf pf[2];
#pragma unroll
    for (int t = 0; t < 2; ++t) pf[t] = frag_ld(&sP[wave][t * 32], 64);
#pragma unroll
    for (int j = 0; j < 4; ++j)
#pragma unroll
      for (int t = 0; t < 2; ++t) {
        v16bf vf = frag_ld(&sVt[cur][(j * 16) * 64 + t * 32], 64);
        oacc[j] = wmma_bf16(pf[t], vf, oacc[j]);
      }

    if (kc + 1 < NC) {
#if HAVE_TDM
      if (wave == 0) wait_tdm();
#else
      wait_cp();
#endif
    }
    __syncthreads();
  }

  // Normalize, bounce O through wave-private LDS, write ctx [B,S,H*DK] with
  // contiguous b128 stores (same-wave DS ops complete in order).
  const int b = bh >> 4, h = bh & 15;
#pragma unroll
  for (int r = 0; r < 8; ++r) {
    float inv = (lrow[r] > 0.0f) ? (1.0f / lrow[r]) : 0.0f;
#pragma unroll
    for (int j = 0; j < 4; ++j)
      sP[wave][(r + hh * 8) * 64 + j * 16 + lr] = (bf16)(oacc[j][r] * inv);
  }
#pragma unroll
  for (int p = 0; p < 4; ++p) {
    int slot = lane + p * 32;              // 0..127
    int row = slot >> 3, c = (slot & 7) << 3;
    v8bf valv = *(const v8bf*)&sP[wave][row * 64 + c];
    *(v8bf*)(ctx + ((size_t)b * SEQ + q0 + wave * 16 + row) * D_MODEL +
             h * DKDIM + c) = valv;
  }
}

// ---------------------------------------------------------------------------
// Launch
// ---------------------------------------------------------------------------
extern "C" void kernel_launch(void* const* d_in, const int* in_sizes, int n_in,
                              void* d_out, int out_size, void* d_ws, size_t ws_size,
                              hipStream_t stream) {
  const float* q  = (const float*)d_in[0];
  const float* k  = (const float*)d_in[1];
  const float* v  = (const float*)d_in[2];
  const float* Wq = (const float*)d_in[3];
  const float* Wk = (const float*)d_in[4];
  const float* Wv = (const float*)d_in[5];
  const float* Wo = (const float*)d_in[6];
  float* out = (float*)d_out;

  const size_t actN = (size_t)MTOT * D_MODEL;    // 8,388,608
  const size_t wN   = (size_t)D_MODEL * D_MODEL; // 1,048,576

  bf16* Xq = (bf16*)d_ws;        // activations in bf16 (16 MB each)
  bf16* Xk = Xq + actN;
  bf16* Xv = Xk + actN;
  bf16* Bq = Xv + actN;          // weights in bf16 (2 MB each)
  bf16* Bk = Bq + wN;
  bf16* Bv = Bk + wN;
  bf16* Bo = Bv + wN;
  bf16* Qp = Bo + wN;            // Q,K: [B,H,S,DK]; V: [B,H,DK,S] (16 MB each)
  bf16* Kp = Qp + actN;
  bf16* VpT = Kp + actN;
  bf16* Cx = VpT + actN;         // attention context, [B,S,D] (16 MB)

  const int cvtThreads = 256;
  mha_cvt_bf16<<<(unsigned)(actN / 4 / cvtThreads), cvtThreads, 0, stream>>>(q, Xq, (int)actN);
  mha_cvt_bf16<<<(unsigned)(actN / 4 / cvtThreads), cvtThreads, 0, stream>>>(k, Xk, (int)actN);
  mha_cvt_bf16<<<(unsigned)(actN / 4 / cvtThreads), cvtThreads, 0, stream>>>(v, Xv, (int)actN);
  mha_cvt_bf16<<<(unsigned)(wN / 4 / cvtThreads),  cvtThreads, 0, stream>>>(Wq, Bq, (int)wN);
  mha_cvt_bf16<<<(unsigned)(wN / 4 / cvtThreads),  cvtThreads, 0, stream>>>(Wk, Bk, (int)wN);
  mha_cvt_bf16<<<(unsigned)(wN / 4 / cvtThreads),  cvtThreads, 0, stream>>>(Wv, Bv, (int)wN);
  mha_cvt_bf16<<<(unsigned)(wN / 4 / cvtThreads),  cvtThreads, 0, stream>>>(Wo, Bo, (int)wN);

  dim3 gemmGrid(D_MODEL / 128, MTOT / 128);  // (8, 64)
  mha_gemm_bt<0><<<gemmGrid, 256, 0, stream>>>(Xq, Bq, (void*)Qp);
  mha_gemm_bt<0><<<gemmGrid, 256, 0, stream>>>(Xk, Bk, (void*)Kp);
  mha_gemm_bt<2><<<gemmGrid, 256, 0, stream>>>(Xv, Bv, (void*)VpT);

  mha_attention<<<dim3(SEQ / 64, BATCH * NHEAD), 128, 0, stream>>>(Qp, Kp, VpT, Cx);

  mha_gemm_bt<1><<<gemmGrid, 256, 0, stream>>>(Cx, Bo, (void*)out);
}
